// Mmoe_layer_42606075577005
// MI455X (gfx1250) — compile-verified, matrix operand order checked by reading, not run on hardware
//
#include <hip/hip_runtime.h>
#include <hip/hip_bf16.h>

typedef __bf16 bf16_t;
typedef __attribute__((ext_vector_type(16))) __bf16 v16bf;
typedef __attribute__((ext_vector_type(8)))  __bf16 v8bf;
typedef __attribute__((ext_vector_type(4)))  __bf16 v4bf;
typedef __attribute__((ext_vector_type(8)))  float  v8f;

#define B_ROWS 16384
#define D_IN   1024
#define H_DIM  256
#define NE     8
#define NT     4

// expert kernel tile
#define M3   32      // B rows per WG
#define NH3  128     // expert_dim cols per WG (2 n-subtiles per wave)
#define KT   64      // K chunk (2 WMMA steps of 32)
#define SA   1032    // A LDS row stride in bf16 (pad +8 -> conflict-free b128)
#define SB   72      // W LDS row stride in bf16 (pad +8)
#define NBUF 3       // weight buffers (issue 2 stages ahead, 1 barrier/stage)
#define NKC  (D_IN / KT)          // 16 K-chunks per expert
#define NST  (NE * NKC)           // 128 total pipeline stages
#define PERW 4       // async-to-LDS instructions per wave per stage

// gate kernel tile
#define MG   64
#define NG   32
#define SAG  1032
#define SBG  72

union Frag { v16bf v; v8bf h[2]; };

__device__ __forceinline__ uint32_t lds_off(const void* p) {
  // gfx1250: low 32 bits of a flat shared address are the LDS offset
  return (uint32_t)(uintptr_t)p;
}

__device__ __forceinline__ void async_b128(uint32_t lds, const void* g) {
  asm volatile("global_load_async_to_lds_b128 %0, %1, off"
               :: "v"(lds), "v"((unsigned long long)(uintptr_t)g) : "memory");
}

// ---------------------------------------------------------------------------
// Kernel 1: We[e][k][n] f32 -> WeT[e][n][k] bf16 (tiled transpose via LDS)
// ---------------------------------------------------------------------------
__global__ __launch_bounds__(256) void convert_We_kernel(
    const float* __restrict__ We, bf16_t* __restrict__ WeT) {
  __shared__ float tile[32][33];
  const int e  = blockIdx.z;
  const int k0 = blockIdx.x * 32;
  const int n0 = blockIdx.y * 32;
  const int tx = threadIdx.x, ty = threadIdx.y;   // 32 x 8
  const float* src = We + (size_t)e * D_IN * H_DIM;
  #pragma unroll
  for (int j = 0; j < 4; ++j)
    tile[ty + 8 * j][tx] = src[(size_t)(k0 + ty + 8 * j) * H_DIM + n0 + tx];
  __syncthreads();
  bf16_t* dst = WeT + (size_t)e * H_DIM * D_IN;
  #pragma unroll
  for (int j = 0; j < 4; ++j)
    dst[(size_t)(n0 + ty + 8 * j) * D_IN + k0 + tx] = (bf16_t)tile[tx][ty + 8 * j];
}

// ---------------------------------------------------------------------------
// Kernel 2: Wg[t][k][e] f32 -> WgT[j=t*8+e][k] bf16
// ---------------------------------------------------------------------------
__global__ __launch_bounds__(256) void convert_Wg_kernel(
    const float* __restrict__ Wg, bf16_t* __restrict__ WgT) {
  const int idx = blockIdx.x * 256 + threadIdx.x;   // 32*1024 total
  const int j = idx >> 10, k = idx & 1023;
  WgT[idx] = (bf16_t)Wg[(size_t)(j >> 3) * (D_IN * NE) + (size_t)k * NE + (j & 7)];
}

// ---------------------------------------------------------------------------
// Kernel 3: gates.  64 rows x 32 logits per WG via WMMA bf16, softmax over
// each group of 8 experts (groups map to 8-lane groups -> shfl_xor reduce).
// ---------------------------------------------------------------------------
__global__ __launch_bounds__(256) void mmoe_gate_kernel(
    const float* __restrict__ x, const bf16_t* __restrict__ WgT,
    const float* __restrict__ bg, float* __restrict__ G) {
  extern __shared__ char smem[];
  bf16_t* Alds = (bf16_t*)smem;                        // [MG][SAG]
  bf16_t* Wlds = (bf16_t*)(smem + MG * SAG * 2);       // [2][NG][SBG]

  const int tid  = threadIdx.x;
  const int row0 = blockIdx.x * MG;

  auto stage_w = [&](int kc, int buf) {
    const int rowN = tid >> 3;                         // 0..31
    const int seg  = tid & 7;                          // 8 x 16B per row
    const bf16_t* g = WgT + ((size_t)rowN * D_IN + kc * KT + seg * 8);
    async_b128(lds_off(Wlds + buf * (NG * SBG) + rowN * SBG + seg * 8), g);
  };

  stage_w(0, 0);  // prologue, overlaps with A staging

  // stage x tile -> bf16 LDS: 64 rows x 256 float4, 64 per thread, coalesced
  const float4* xv = (const float4*)(x + (size_t)row0 * D_IN);
  #pragma unroll
  for (int i = 0; i < 64; ++i) {
    const int idx = tid + 256 * i;
    const int r = idx >> 8, kq = idx & 255;
    float4 f = xv[idx];
    v4bf b4 = {(bf16_t)f.x, (bf16_t)f.y, (bf16_t)f.z, (bf16_t)f.w};
    *(v4bf*)(Alds + r * SAG + kq * 4) = b4;
  }
  __syncthreads();

  const int lane = tid & 31, w = tid >> 5;
  const int m0 = (w >> 1) * 16, n0 = (w & 1) * 16;
  const int half = lane >> 4, l16 = lane & 15;

  v8f acc = {0.f, 0.f, 0.f, 0.f, 0.f, 0.f, 0.f, 0.f};
  int sbuf = 0;
  for (int kc = 0; kc < D_IN / KT; ++kc) {
    if (kc < D_IN / KT - 1) {
      stage_w(kc + 1, sbuf ^ 1);
      asm volatile("s_wait_asynccnt 0x1" ::: "memory");
    } else {
      asm volatile("s_wait_asynccnt 0x0" ::: "memory");
    }
    __syncthreads();
    const bf16_t* Wb = Wlds + sbuf * (NG * SBG);
    #pragma unroll
    for (int s2 = 0; s2 < 2; ++s2) {
      const int kL = kc * KT + s2 * 32 + half * 8;
      Frag a, b;
      const bf16_t* ap = Alds + (m0 + l16) * SAG + kL;
      a.h[0] = *(const v8bf*)ap;
      a.h[1] = *(const v8bf*)(ap + 16);
      const bf16_t* bp = Wb + (n0 + l16) * SBG + s2 * 32 + half * 8;
      b.h[0] = *(const v8bf*)bp;
      b.h[1] = *(const v8bf*)(bp + 16);
      acc = __builtin_amdgcn_wmma_f32_16x16x32_bf16(false, a.v, false, b.v,
                                                    (short)0, acc, false, false);
    }
    __syncthreads();
    sbuf ^= 1;
  }

  const int j = n0 + l16;
  const float bgv = bg[j];
  #pragma unroll
  for (int r = 0; r < 8; ++r) {
    float lg = acc[r] + bgv;
    float m = lg;
    m = fmaxf(m, __shfl_xor(m, 1));
    m = fmaxf(m, __shfl_xor(m, 2));
    m = fmaxf(m, __shfl_xor(m, 4));
    float ex = __expf(lg - m);
    float s = ex;
    s += __shfl_xor(s, 1);
    s += __shfl_xor(s, 2);
    s += __shfl_xor(s, 4);
    const int rr = m0 + r + 8 * half;                 // 0..63
    G[(size_t)(row0 + rr) * 32 + j] = ex / s;
  }
}

// ---------------------------------------------------------------------------
// Kernel 4: fused experts + ReLU + gated combine.
// WG tile: 32 B-rows x 128 h-cols, loops all 8 experts. x tile resident in
// LDS (bf16). Weight K-chunks triple-buffered via async-to-LDS, issued two
// stages ahead -> single barrier per stage, 4 WMMAs per wave per barrier.
// ---------------------------------------------------------------------------
__global__ __launch_bounds__(256) void mmoe_expert_kernel(
    const float* __restrict__ x, const bf16_t* __restrict__ WeT,
    const float* __restrict__ be, const float* __restrict__ G,
    float* __restrict__ out) {
  extern __shared__ char smem[];
  bf16_t* Alds  = (bf16_t*)smem;                                   // [M3][SA]
  bf16_t* Wlds  = (bf16_t*)(smem + M3 * SA * 2);                   // [NBUF][NH3][SB]
  float*  Glds  = (float*)(smem + M3 * SA * 2 + NBUF * NH3 * SB * 2); // [M3][32]
  float*  beLds = Glds + M3 * 32;                                  // [NE][NH3]

  const int tid   = threadIdx.x;
  const int row0  = blockIdx.x * M3;
  const int hbase = blockIdx.y * NH3;

  // stage s = e*NKC + kc ; each thread issues 4 b128 async copies
  auto stage_w = [&](int s, int buf) {
    const int e  = s >> 4;        // s / NKC
    const int kc = s & 15;        // s % NKC
    #pragma unroll
    for (int i = 0; i < 4; ++i) {                  // 1024 x 16B transfers
      const int id   = tid * 4 + i;
      const int rowN = id >> 3;                    // 0..127
      const int seg  = id & 7;
      const bf16_t* g =
          WeT + (((size_t)e * H_DIM + hbase + rowN) * D_IN + kc * KT + seg * 8);
      async_b128(lds_off(Wlds + buf * (NH3 * SB) + rowN * SB + seg * 8), g);
    }
  };

  stage_w(0, 0);   // prologue: two stages in flight
  stage_w(1, 1);

  // stage x tile -> bf16: 32 rows x 256 float4, 32 per thread
  const float4* xv = (const float4*)(x + (size_t)row0 * D_IN);
  #pragma unroll
  for (int i = 0; i < 32; ++i) {
    const int idx = tid + 256 * i;
    const int r = idx >> 8, kq = idx & 255;
    float4 f = xv[idx];
    v4bf b4 = {(bf16_t)f.x, (bf16_t)f.y, (bf16_t)f.z, (bf16_t)f.w};
    *(v4bf*)(Alds + r * SA + kq * 4) = b4;
  }
  // stage gates (32x32) and bias (8x128)
  #pragma unroll
  for (int i = 0; i < 4; ++i) {
    const int idx = tid + 256 * i;
    Glds[idx] = G[(size_t)(row0 + (idx >> 5)) * 32 + (idx & 31)];
  }
  #pragma unroll
  for (int i = 0; i < 4; ++i) {
    const int idx = tid + 256 * i;
    beLds[idx] = be[(idx >> 7) * H_DIM + hbase + (idx & 127)];
  }

  const int lane = tid & 31, w = tid >> 5;
  const int m0 = (w >> 2) * 16;        // 0 / 16
  const int n0 = (w & 3) * 32;         // wave owns cols [n0, n0+32)
  const int half = lane >> 4, l16 = lane & 15;

  float tw[NT][2][8];
  #pragma unroll
  for (int t = 0; t < NT; ++t)
    #pragma unroll
    for (int q = 0; q < 2; ++q)
      #pragma unroll
      for (int r = 0; r < 8; ++r) tw[t][q][r] = 0.f;

  for (int e = 0; e < NE; ++e) {
    v8f acc0 = {0.f, 0.f, 0.f, 0.f, 0.f, 0.f, 0.f, 0.f};
    v8f acc1 = {0.f, 0.f, 0.f, 0.f, 0.f, 0.f, 0.f, 0.f};
    for (int kc = 0; kc < NKC; ++kc) {
      const int s = e * NKC + kc;
      // stage s complete when at most stage s+1's ops remain outstanding
      if (s < NST - 1) asm volatile("s_wait_asynccnt 0x4" ::: "memory");
      else             asm volatile("s_wait_asynccnt 0x0" ::: "memory");
      __syncthreads();  // data-ready + WAR guard for the buffer refilled below
      if (s + 2 < NST) stage_w(s + 2, (s + 2) % NBUF);

      const bf16_t* Wb = Wlds + (s % NBUF) * (NH3 * SB);
      #pragma unroll
      for (int s2 = 0; s2 < 2; ++s2) {
        const int kL = kc * KT + s2 * 32 + half * 8;
        Frag a, b0, b1;
        const bf16_t* ap = Alds + (m0 + l16) * SA + kL;
        a.h[0] = *(const v8bf*)ap;
        a.h[1] = *(const v8bf*)(ap + 16);
        const bf16_t* bp0 = Wb + (n0 + l16) * SB + s2 * 32 + half * 8;
        b0.h[0] = *(const v8bf*)bp0;
        b0.h[1] = *(const v8bf*)(bp0 + 16);
        const bf16_t* bp1 = bp0 + 16 * SB;
        b1.h[0] = *(const v8bf*)bp1;
        b1.h[1] = *(const v8bf*)(bp1 + 16);
        acc0 = __builtin_amdgcn_wmma_f32_16x16x32_bf16(false, a.v, false, b0.v,
                                                       (short)0, acc0, false, false);
        acc1 = __builtin_amdgcn_wmma_f32_16x16x32_bf16(false, a.v, false, b1.v,
                                                       (short)0, acc1, false, false);
      }
    }
    // bias + ReLU + gated tower accumulation
    const float bias0 = beLds[e * NH3 + n0 + l16];
    const float bias1 = beLds[e * NH3 + n0 + 16 + l16];
    #pragma unroll
    for (int r = 0; r < 8; ++r) {
      float v0 = acc0[r] + bias0;
      float v1 = acc1[r] + bias1;
      v0 = v0 > 0.f ? v0 : 0.f;
      v1 = v1 > 0.f ? v1 : 0.f;
      const int rr = m0 + r + 8 * half;            // row in tile, 0..31
      #pragma unroll
      for (int t = 0; t < NT; ++t) {
        const float g = Glds[rr * 32 + t * 8 + e]; // LDS broadcast read
        tw[t][0][r] += g * v0;
        tw[t][1][r] += g * v1;
      }
    }
  }

  // write towers[t][b][h]
  #pragma unroll
  for (int t = 0; t < NT; ++t)
    #pragma unroll
    for (int r = 0; r < 8; ++r) {
      const int rr = m0 + r + 8 * half;
      const size_t base = ((size_t)t * B_ROWS + row0 + rr) * H_DIM + hbase;
      out[base + n0 + l16]      = tw[t][0][r];
      out[base + n0 + 16 + l16] = tw[t][1][r];
    }
}

// ---------------------------------------------------------------------------
extern "C" void kernel_launch(void* const* d_in, const int* in_sizes, int n_in,
                              void* d_out, int out_size, void* d_ws, size_t ws_size,
                              hipStream_t stream) {
  (void)in_sizes; (void)n_in; (void)out_size; (void)ws_size;
  const float* x  = (const float*)d_in[0];
  const float* We = (const float*)d_in[1];
  const float* be = (const float*)d_in[2];
  const float* Wg = (const float*)d_in[3];
  const float* bg = (const float*)d_in[4];
  float* out = (float*)d_out;

  char* ws = (char*)d_ws;
  bf16_t* WeT = (bf16_t*)ws;                                   // 4 MB
  const size_t WET_BYTES = (size_t)NE * H_DIM * D_IN * 2;
  bf16_t* WgT = (bf16_t*)(ws + WET_BYTES);                     // 64 KB
  const size_t WGT_BYTES = (size_t)NG * D_IN * 2;
  float* G = (float*)(ws + WET_BYTES + WGT_BYTES);             // 2 MB

  convert_We_kernel<<<dim3(D_IN / 32, H_DIM / 32, NE), dim3(32, 8), 0, stream>>>(We, WeT);
  convert_Wg_kernel<<<dim3((NG * D_IN) / 256), dim3(256), 0, stream>>>(Wg, WgT);

  const size_t gate_lds = (size_t)MG * SAG * 2 + 2 * NG * SBG * 2;
  mmoe_gate_kernel<<<dim3(B_ROWS / MG), dim3(256), gate_lds, stream>>>(x, WgT, bg, G);

  const size_t exp_lds = (size_t)M3 * SA * 2 + (size_t)NBUF * NH3 * SB * 2 +
                         (size_t)M3 * 32 * 4 + (size_t)NE * NH3 * 4;   // 129,536 B
  mmoe_expert_kernel<<<dim3(B_ROWS / M3, H_DIM / NH3), dim3(256), exp_lds, stream>>>(
      x, WeT, be, G, out);
}
